// DenseGATBlock_77421080477768
// MI455X (gfx1250) — matrix operand order repaired and checked by reading, not compiled, not run on hardware
//
#include <hip/hip_runtime.h>
#include <hip/hip_bf16.h>

typedef __attribute__((ext_vector_type(16))) __bf16 v16bf;
typedef __attribute__((ext_vector_type(8)))  float  v8f;

#define D 128
#define NEG_SLOPE 0.2f

// ---------------- Kernel 1: adj (f32, 256MB) -> bitmask (8MB), read adj from HBM once
__global__ void mask_kernel(const float* __restrict__ adj, unsigned* __restrict__ bits,
                            int nwords) {
  int w = blockIdx.x * blockDim.x + threadIdx.x;
  if (w >= nwords) return;
  const float* p = adj + (size_t)w * 32;
  unsigned m = 0;
#pragma unroll
  for (int b = 0; b < 32; ++b) m |= (p[b] != 0.0f ? 1u : 0u) << b;
  bits[w] = m;
}

// ---------------- Kernel 2: H = X@W via v_wmma_f32_16x16x32_bf16.
// Writes H (f32, row-major) and HbT (bf16, transposed [D][N]) for the aggregation GEMM.
__global__ void gemm_h_kernel(const float* __restrict__ X, const float* __restrict__ W,
                              float* __restrict__ H, __bf16* __restrict__ HbT, int N) {
  int wid  = (blockIdx.x * blockDim.x + threadIdx.x) >> 5;
  int lane = threadIdx.x & 31;
  int it = (wid >> 3) * 16;        // 16-row tile of X
  int ct = (wid & 7) * 16;         // 16-col tile of W
  if (it >= N) return;
  int m = lane & 15, half = lane >> 4;

  v8f acc = {0.f, 0.f, 0.f, 0.f, 0.f, 0.f, 0.f, 0.f};
  const float* xrow = X + (size_t)(it + m) * D;

#pragma unroll
  for (int ks = 0; ks < D; ks += 32) {
    v16bf a, b;
#pragma unroll
    for (int v = 0; v < 8; ++v) {
      // A 16x32 bf16, contiguous k order used here: lanes 0-15 M=lane / 16-31 M=lane-16;
      // VGPR v: K = 16*half + 2v (+e)
      int kb = 16 * half + 2 * v;
      a[2 * v]     = (__bf16)xrow[ks + kb];
      a[2 * v + 1] = (__bf16)xrow[ks + kb + 1];
      // B 32x16 bf16, same contiguous k order as A: lane = N column, K = 16*half + 2v (+e)
      int kk = 16 * half + 2 * v;
      b[2 * v]     = (__bf16)W[(size_t)(ks + kk) * D + ct + m];
      b[2 * v + 1] = (__bf16)W[(size_t)(ks + kk + 1) * D + ct + m];
    }
    acc = __builtin_amdgcn_wmma_f32_16x16x32_bf16(false, a, false, b, (short)0, acc,
                                                  false, false);
  }

#pragma unroll
  for (int r = 0; r < 8; ++r) {
    // C/D f32 16x16 layout: VGPR r -> M = r + 8*half, N = lane%16
    int row = it + r + 8 * half;
    int col = ct + m;
    H[(size_t)row * D + col]   = acc[r];
    HbT[(size_t)col * N + row] = (__bf16)acc[r];
  }
}

// ---------------- Kernel 3: a_src[i] = H[i,:].att_src, a_dst[i] = H[i,:].att_dst (wave per row)
__global__ void att_kernel(const float* __restrict__ H, const float* __restrict__ asv,
                           const float* __restrict__ adv, float* __restrict__ a_src,
                           float* __restrict__ a_dst, int N) {
  int row  = (blockIdx.x * blockDim.x + threadIdx.x) >> 5;
  int lane = threadIdx.x & 31;
  if (row >= N) return;
  const float* h = H + (size_t)row * D;
  float s1 = 0.f, s2 = 0.f;
#pragma unroll
  for (int c = lane; c < D; c += 32) {
    float v = h[c];
    s1 += v * asv[c];
    s2 += v * adv[c];
  }
#pragma unroll
  for (int off = 16; off > 0; off >>= 1) {
    s1 += __shfl_xor(s1, off, 32);
    s2 += __shfl_xor(s2, off, 32);
  }
  if (lane == 0) { a_src[row] = s1; a_dst[row] = s2; }
}

// ---------------- Kernel 4: per-row masked max + softmax denominator (bitmask scan, ctz loop).
// leaky_relu is monotone -> M_i = leaky(max_masked a_src[j] + a_dst[i]).
__global__ void stats_kernel(const unsigned* __restrict__ bits, const float* __restrict__ a_src,
                             const float* __restrict__ a_dst, float* __restrict__ Mrow,
                             float* __restrict__ den, int N) {
  int row  = (blockIdx.x * blockDim.x + threadIdx.x) >> 5;
  int lane = threadIdx.x & 31;
  if (row >= N) return;
  int nw = N >> 5;
  const unsigned* brow = bits + (size_t)row * nw;

  float mx = -3.0e38f;
  for (int w = lane; w < nw; w += 32) {
    unsigned bw = brow[w];
    int j0 = w << 5;
    while (bw) {
      int b = __builtin_ctz(bw);
      bw &= bw - 1;
      mx = fmaxf(mx, a_src[j0 + b]);
    }
  }
#pragma unroll
  for (int off = 16; off > 0; off >>= 1) mx = fmaxf(mx, __shfl_xor(mx, off, 32));

  float adst = a_dst[row];
  float t  = mx + adst;
  float Mi = t >= 0.f ? t : NEG_SLOPE * t;

  float s = 0.f;
  for (int w = lane; w < nw; w += 32) {
    unsigned bw = brow[w];
    int j0 = w << 5;
    while (bw) {
      int b = __builtin_ctz(bw);
      bw &= bw - 1;
      float e = a_src[j0 + b] + adst;
      e = e >= 0.f ? e : NEG_SLOPE * e;
      s += __expf(e - Mi);
    }
  }
#pragma unroll
  for (int off = 16; off > 0; off >>= 1) s += __shfl_xor(s, off, 32);

  if (lane == 0) { Mrow[row] = Mi; den[row] = s; }
}

// ---------------- Kernel 5: flash-style numerator num[i,:] = sum_j w_ij * Hb[j,:]
// w-tile (16x32) built in registers as WMMA A operand, contiguous k = 16*half + s; 8 WMMAs per j-tile cover c=0..127.
// Each wave: one 16-row i-tile, one of NCHUNK j-ranges; writes a disjoint f32 partial.
__global__ void agg_kernel(const unsigned* __restrict__ bits, const float* __restrict__ a_src,
                           const float* __restrict__ a_dst, const float* __restrict__ Mrow,
                           const __bf16* __restrict__ HbT, float* __restrict__ numPart,
                           int N, int nchunks) {
  int wid  = (blockIdx.x * blockDim.x + threadIdx.x) >> 5;
  int lane = threadIdx.x & 31;
  int it = (wid / nchunks) * 16;
  int chunk = wid % nchunks;
  if (it >= N) return;
  int m = lane & 15, half = lane >> 4;
  int row = it + m;
  float adst = a_dst[row];
  float Mi   = Mrow[row];
  int jspan = N / nchunks;
  int jbeg = chunk * jspan, jend = jbeg + jspan;
  const unsigned* brow = bits + (size_t)row * (N >> 5);

  v8f acc[8];
  v8f z = {0.f, 0.f, 0.f, 0.f, 0.f, 0.f, 0.f, 0.f};
#pragma unroll
  for (int c = 0; c < 8; ++c) acc[c] = z;

  for (int jt = jbeg; jt < jend; jt += 32) {
    unsigned mw = brow[jt >> 5];
    v16bf aT;
#pragma unroll
    for (int v = 0; v < 8; ++v) {
#pragma unroll
      for (int e = 0; e < 2; ++e) {
        int K = 16 * half + 2 * v + e;
        float wv = 0.f;
        if ((mw >> K) & 1u) {
          float ei = a_src[jt + K] + adst;
          ei = ei >= 0.f ? ei : NEG_SLOPE * ei;
          wv = __expf(ei - Mi);            // <= 1.0, safe in bf16
        }
        aT[2 * v + e] = (__bf16)wv;
      }
    }
#pragma unroll
    for (int c = 0; c < 8; ++c) {
      // B tile: HbT[c*16 + n][jt + 16*half .. +15]  -> 32B contiguous per lane (b128 loads)
      const v16bf* bp = (const v16bf*)(HbT + (size_t)(c * 16 + m) * N + jt + 16 * half);
      v16bf bT = *bp;
      acc[c] = __builtin_amdgcn_wmma_f32_16x16x32_bf16(false, aT, false, bT, (short)0,
                                                       acc[c], false, false);
    }
  }

  float* out = numPart + (size_t)chunk * N * D;
#pragma unroll
  for (int c = 0; c < 8; ++c)
#pragma unroll
    for (int r = 0; r < 8; ++r)
      out[(size_t)(it + r + 8 * half) * D + c * 16 + m] = acc[c][r];
}

// ---------------- Kernel 6: out = sum(partials)/den + bias (+relu for layer 0)
__global__ void finalize_kernel(const float* __restrict__ numPart, const float* __restrict__ den,
                                const float* __restrict__ bias, float* __restrict__ out,
                                int N, int nchunks, int do_relu) {
  int idx = blockIdx.x * blockDim.x + threadIdx.x;
  int total = N * D;
  if (idx >= total) return;
  int row = idx >> 7, col = idx & 127;
  float s = 0.f;
  for (int c = 0; c < nchunks; ++c) s += numPart[(size_t)c * total + idx];
  s = s / den[row] + bias[col];
  if (do_relu) s = fmaxf(s, 0.f);
  out[idx] = s;
}

extern "C" void kernel_launch(void* const* d_in, const int* in_sizes, int n_in,
                              void* d_out, int out_size, void* d_ws, size_t ws_size,
                              hipStream_t stream) {
  (void)n_in; (void)out_size; (void)ws_size;
  const float* x   = (const float*)d_in[0];
  const float* adj = (const float*)d_in[1];
  const float* Wl[2]  = {(const float*)d_in[2], (const float*)d_in[6]};
  const float* asv[2] = {(const float*)d_in[3], (const float*)d_in[7]};
  const float* adv[2] = {(const float*)d_in[4], (const float*)d_in[8]};
  const float* bl[2]  = {(const float*)d_in[5], (const float*)d_in[9]};

  const int N = in_sizes[0] / D;   // 8192
  const int NCHUNK = 4;

  char* ws = (char*)d_ws;
  size_t off = 0;
  auto carve = [&](size_t bytes) {
    char* p = ws + off;
    off = (off + bytes + 255) & ~(size_t)255;
    return p;
  };
  unsigned* bits  = (unsigned*)carve((size_t)N * N / 8);          // 8 MB, L2-resident
  float*    H     = (float*)carve((size_t)N * D * 4);             // 4 MB
  __bf16*   HbT   = (__bf16*)carve((size_t)N * D * 2);            // 2 MB
  float*    a_src = (float*)carve((size_t)N * 4);
  float*    a_dst = (float*)carve((size_t)N * 4);
  float*    Mr    = (float*)carve((size_t)N * 4);
  float*    den   = (float*)carve((size_t)N * 4);
  float*    numP  = (float*)carve((size_t)NCHUNK * N * D * 4);    // 16 MB

  int nwords = (N * N) >> 5;
  mask_kernel<<<(nwords + 255) / 256, 256, 0, stream>>>(adj, bits, nwords);

  float* out0 = (float*)d_out;
  float* out1 = out0 + (size_t)N * D;

  for (int layer = 0; layer < 2; ++layer) {
    const float* X = (layer == 0) ? x : out0;   // out0 already has bias+relu applied
    float* out = (layer == 0) ? out0 : out1;

    int gemmWaves = (N / 16) * (D / 16);
    gemm_h_kernel<<<(gemmWaves * 32 + 255) / 256, 256, 0, stream>>>(X, Wl[layer], H, HbT, N);
    att_kernel<<<(N * 32 + 255) / 256, 256, 0, stream>>>(H, asv[layer], adv[layer],
                                                         a_src, a_dst, N);
    stats_kernel<<<(N * 32 + 255) / 256, 256, 0, stream>>>(bits, a_src, a_dst, Mr, den, N);
    int aggWaves = (N / 16) * NCHUNK;
    agg_kernel<<<(aggWaves * 32 + 255) / 256, 256, 0, stream>>>(bits, a_src, a_dst, Mr,
                                                                HbT, numP, N, NCHUNK);
    finalize_kernel<<<(N * D + 255) / 256, 256, 0, stream>>>(numP, den, bl[layer], out,
                                                             N, NCHUNK, layer == 0 ? 1 : 0);
  }
}
